// myLoss_73598559584710
// MI455X (gfx1250) — compile-verified, matrix operand order checked by reading, not run on hardware
//
#include <hip/hip_runtime.h>

typedef __attribute__((ext_vector_type(8))) int v8i;

#define N_ELEM          16777216u   // 4096*4096
#define BLOCK           256
#define WAVES_PER_BLOCK 8
#define GRID            512
#define ITERS           4
// per wave-iteration: 8 j-steps * 32 lanes * 4 elem = 1024 elements -> one V_WMMA_I32_16X16X64_IU8
// GRID * WAVES_PER_BLOCK * ITERS * 1024 = 16,777,216 = N  (exact, no tail, EXEC all-ones)

__global__ void zero_ws_kernel(unsigned int* ws) { ws[0] = 0u; }

__device__ __forceinline__ unsigned int weight_byte(float c0, float c1, int l) {
    // argmax over 2 classes after (monotonic) softmax: class1 wins iff strictly greater
    bool pm1 = c1 > c0;
    // label==1 & predicted 0 -> 3 ; label==0 & predicted 1 -> 1 ; else 0
    return l ? (pm1 ? 0u : 3u) : (pm1 ? 1u : 0u);
}

__global__ __launch_bounds__(BLOCK) void loss_reduce_kernel(
        const float* __restrict__ pred,
        const int*   __restrict__ label,
        unsigned int* __restrict__ ws)
{
    const float* __restrict__ p0 = pred;            // class 0 plane
    const float* __restrict__ p1 = pred + N_ELEM;   // class 1 plane

    const int lane = threadIdx.x & 31;
    const int wave = threadIdx.x >> 5;

    v8i acc = {0, 0, 0, 0, 0, 0, 0, 0};
    v8i bones;
#pragma unroll
    for (int i = 0; i < 8; ++i) bones[i] = 0x01010101;   // B = all-ones u8 matrix

    const unsigned int waveBase =
        (blockIdx.x * WAVES_PER_BLOCK + wave) * (1024u * ITERS);

#pragma unroll
    for (int it = 0; it < ITERS; ++it) {
        const unsigned int base = waveBase + (unsigned int)it * 1024u;
        v8i a;
#pragma unroll
        for (int j = 0; j < 8; ++j) {
            const unsigned int idx = base + (unsigned int)j * 128u + (unsigned int)lane * 4u;
            const float4 c0 = *(const float4*)(p0 + idx);     // global_load_b128, coalesced 512B/wave
            const float4 c1 = *(const float4*)(p1 + idx);
            const int4   lb = *(const int4*)(label + idx);
            unsigned int w0 = weight_byte(c0.x, c1.x, lb.x);
            unsigned int w1 = weight_byte(c0.y, c1.y, lb.y);
            unsigned int w2 = weight_byte(c0.z, c1.z, lb.z);
            unsigned int w3 = weight_byte(c0.w, c1.w, lb.w);
            a[j] = (int)(w0 | (w1 << 8) | (w2 << 16) | (w3 << 24));
        }
        // D[m][n] += sum_k A[m][k] * 1  -> every packed weight byte counted exactly once
        // (summed D tile = 16 * total, independent of byte->(m,k) mapping)
        acc = __builtin_amdgcn_wmma_i32_16x16x64_iu8(
                  /*sgn_a=*/false, a, /*sgn_b=*/false, bones, acc,
                  /*reuse_a=*/false, /*reuse_b=*/false);
    }

    // Sum the 8 accumulator VGPRs held by this lane, then wave32 xor-shuffle reduce.
    int s = acc[0] + acc[1] + acc[2] + acc[3] + acc[4] + acc[5] + acc[6] + acc[7];
#pragma unroll
    for (int off = 16; off > 0; off >>= 1)
        s += __shfl_xor(s, off, 32);

    // s == 16 * (this wave's weight total); exact integer divide, one atomic per wave.
    if (lane == 0)
        atomicAdd(ws, (unsigned int)(s >> 4));
}

__global__ void finalize_kernel(const unsigned int* __restrict__ ws,
                                float* __restrict__ out)
{
    // total <= 3*N = 50,331,648 : exact in double, then round once to f32
    out[0] = (float)((double)ws[0] / (double)N_ELEM);
}

extern "C" void kernel_launch(void* const* d_in, const int* in_sizes, int n_in,
                              void* d_out, int out_size, void* d_ws, size_t ws_size,
                              hipStream_t stream)
{
    const float* pred  = (const float*)d_in[0];   // (1, 2, 4096, 4096) f32
    const int*   label = (const int*)d_in[1];     // (1, 4096, 4096) i32
    unsigned int* ws   = (unsigned int*)d_ws;
    float*        out  = (float*)d_out;

    zero_ws_kernel<<<1, 1, 0, stream>>>(ws);
    loss_reduce_kernel<<<GRID, BLOCK, 0, stream>>>(pred, label, ws);
    finalize_kernel<<<1, 1, 0, stream>>>(ws, out);
}
